// QuantMLP_33019708572175
// MI455X (gfx1250) — matrix operand order repaired and checked by reading, not compiled, not run on hardware
//
#include <hip/hip_runtime.h>

typedef __attribute__((ext_vector_type(16))) _Float16 v16h;
typedef __attribute__((ext_vector_type(8)))  _Float16 v8h;
typedef __attribute__((ext_vector_type(8)))  float    v8f;
typedef __attribute__((ext_vector_type(4)))  unsigned int u32x4;
typedef __attribute__((ext_vector_type(8)))  int      i32x8;
typedef __attribute__((ext_vector_type(4)))  int      i32x4;

#define BATCH   131072
#define NIN     784
#define K1P     800            // 784 padded to 25 * 32 (pad cols of W1 are zero)
#define H1      128
#define H2      64
#define NOUT    10
#define NOUTP   16
#define ROWS_WG 128            // 8 waves * 16 rows
#define NTILES  (BATCH / ROWS_WG)   // 1024
#define NWG     256

// x staging: per wave, 5 chunks of 5 k-steps; chunk = 16 rows x 160 halves (TDM tile)
#define CHUNK_W   160                    // halves per row per chunk
#define CHUNK_H   (16 * CHUNK_W)         // 2560 halves = 5120 bytes
#define STAGE_H   (2 * CHUNK_H)          // 5120 halves = 10240 bytes per wave

// LDS layout (units: halves)
#define W1_OFF  0                        // [128][800]  = 102400 halves
#define W2_OFF  (H1 * K1P)               // [64][128]   -> 102400
#define W3_OFF  (W2_OFF + H2 * H1)       // [16][64]    -> 110592
#define STG_OFF (W3_OFF + NOUTP * H2)    // 8 * 5120    -> 111616
#define H2R_OFF (STG_OFF + 8 * STAGE_H)  // 8 * 1024    -> 152576 (dedicated h2)
#define LDS_HALVES (H2R_OFF + 8 * 16 * H2) // 160768 halves = 321536 B (< 327680)

extern __shared__ _Float16 smem[];

union AFrag { v16h v; v8h h[2]; };

__device__ __forceinline__ v8f wmma16(const v16h& a, const v16h& b, const v8f& c) {
    return __builtin_amdgcn_wmma_f32_16x16x32_f16(
        false, a, false, b, (short)0, c, false, false);
}

// Issue one TDM transfer: 16 rows x 160 halves of x (row stride 784 halves)
// into LDS at lds_byte (packed [16][160] halves). D# per CDNA5 ISA section 8.
__device__ __forceinline__ void tdm_load_chunk(const _Float16* x, size_t elem_off,
                                               unsigned lds_byte) {
    unsigned long long ga = (unsigned long long)(uintptr_t)x + 2ull * elem_off;
    u32x4 g0;
    g0[0] = 1u;                                   // count=1 (valid), no gather
    g0[1] = lds_byte;                             // lds_addr (bytes)
    g0[2] = (unsigned)(ga & 0xFFFFFFFFu);         // global_addr[31:0]
    g0[3] = (unsigned)((ga >> 32) & 0x01FFFFFFu)  // global_addr[56:32]
          | (2u << 30);                           // type=2 ("image")
    i32x8 g1;
    g1[0] = (int)(1u << 16);                      // data_size=1 -> 2 bytes/elem
    g1[1] = (int)(784u << 16);                    // tensor_dim0 = 784 (low16)
    g1[2] = 0;                                    // dim0 hi, tensor_dim1 low16 (131072)
    g1[3] = (int)(2u | ((unsigned)CHUNK_W << 16));// tensor_dim1 hi16=2, tile_dim0=160
    g1[4] = 16;                                   // tile_dim1=16, tile_dim2=0
    g1[5] = 784;                                  // tensor_dim0_stride low32
    g1[6] = 0;                                    // stride hi, dim1_stride lo
    g1[7] = 0;
    i32x4 gz4 = {0, 0, 0, 0};                     // groups 2/3: dims 2..4 unused
    i32x8 gz8 = {0, 0, 0, 0, 0, 0, 0, 0};         // extra operand (probe: zero ok)
    __builtin_amdgcn_tensor_load_to_lds(g0, g1, gz4, gz4, gz8, 0);
}

extern "C" __global__ __launch_bounds__(256, 1)
void qmlp_fused_w4a16(const _Float16* __restrict__ x,
                      const int*      __restrict__ w1q,
                      const _Float16* __restrict__ s1,
                      const _Float16* __restrict__ b1,
                      const int*      __restrict__ w2q,
                      const _Float16* __restrict__ s2,
                      const _Float16* __restrict__ b2,
                      const int*      __restrict__ w3q,
                      const _Float16* __restrict__ s3,
                      const _Float16* __restrict__ b3,
                      _Float16*       __restrict__ out)
{
    const int tid  = threadIdx.x;
    const int lane = tid & 31;
    const int wave = tid >> 5;
    const int l15  = lane & 15;
    const int hi   = lane >> 4;
    const int kbA  = hi ? 8 : 0;      // A-frag K sub-offset (16-bit A layout)
    const int kbB  = hi ? 16 : 0;     // B-frag K sub-offset (contiguous k = 16*hi + s)
    const int rowoff = hi * 8;        // C/D frag: lanes>=16 hold rows M=8..15

    // ---- one-time cooperative weight dequant into LDS (f16 math == reference) ----
    for (int i = tid; i < H1 * K1P; i += 256) {
        int n = i / K1P, k = i - n * K1P;
        _Float16 v = (_Float16)0.0f;
        if (k < NIN) v = (_Float16)(float)w1q[n * NIN + k] * s1[n];
        smem[W1_OFF + i] = v;                               // [n][k], K padded with 0
    }
    for (int i = tid; i < H2 * H1; i += 256) {
        int n = i >> 7;
        smem[W2_OFF + i] = (_Float16)(float)w2q[i] * s2[n];
    }
    for (int i = tid; i < NOUTP * H2; i += 256) {
        int n = i >> 6, k = i & 63;
        _Float16 v = (_Float16)0.0f;
        if (n < NOUT) v = (_Float16)(float)w3q[n * H2 + k] * s3[n];
        smem[W3_OFF + i] = v;                               // rows 10..15 zero
    }
    __syncthreads();

    // Hoisted per-lane bias columns
    float bb1[8], bb2[4];
    #pragma unroll
    for (int n = 0; n < 8; ++n) bb1[n] = (float)b1[n * 16 + l15];
    #pragma unroll
    for (int n = 0; n < 4; ++n) bb2[n] = (float)b2[n * 16 + l15];
    const float bb3 = (l15 < NOUT) ? (float)b3[l15] : 0.0f;

    const int      stage_h = STG_OFF + wave * STAGE_H;         // halves
    const unsigned stage_b = (unsigned)(2 * stage_h);          // bytes
    _Float16* h2w = &smem[H2R_OFF + wave * 16 * H2];           // dedicated [16][64]

    // Prologue: first tile's chunk 0 into buffer 0 (tc=0 parity).
    {
        const int t0 = blockIdx.x;
        if (t0 < NTILES)
            tdm_load_chunk(x, (size_t)(t0 * ROWS_WG + wave * 16) * NIN, stage_b);
    }

    int tc = 0;  // local tile counter: chunk c of tile tc lives in buffer (tc+c)&1
    for (int tile = blockIdx.x; tile < NTILES; tile += gridDim.x, ++tc) {
        const int    m0   = tile * ROWS_WG + wave * 16;
        const size_t base = (size_t)m0 * NIN;

        // ---------------- layer 1: [16x784] x [784x128], TDM-staged A ----------
        v8f acc[8];
        #pragma unroll
        for (int n = 0; n < 8; ++n) acc[n] = {};

        for (int c = 0; c < 5; ++c) {
            const int curbuf = (tc + c) & 1;
            if (c < 4) {
                tdm_load_chunk(x, base + (size_t)(c + 1) * CHUNK_W,
                               stage_b + (unsigned)((curbuf ^ 1) * 2 * CHUNK_H));
                __builtin_amdgcn_s_wait_tensorcnt(1);           // chunk c landed
            } else {
                const int ntile = tile + (int)gridDim.x;
                if (ntile < NTILES) {
                    // overlap next tile's first chunk with L2/L3/epilogue
                    tdm_load_chunk(x, (size_t)(ntile * ROWS_WG + wave * 16) * NIN,
                                   stage_b + (unsigned)((curbuf ^ 1) * 2 * CHUNK_H));
                    __builtin_amdgcn_s_wait_tensorcnt(1);
                } else {
                    __builtin_amdgcn_s_wait_tensorcnt(0);
                }
            }
            asm volatile("" ::: "memory");                      // fence LDS reads
            const _Float16* xs = &smem[stage_h + curbuf * CHUNK_H];
            #pragma unroll
            for (int j = 0; j < 5; ++j) {
                AFrag a;
                a.h[0] = *(const v8h*)&xs[l15 * CHUNK_W + j * 32 + kbB];
                a.h[1] = *(const v8h*)&xs[l15 * CHUNK_W + j * 32 + kbB + 8];
                const int k0 = c * CHUNK_W + j * 32;            // global K offset
                #pragma unroll
                for (int n = 0; n < 8; ++n) {
                    const v16h b = *(const v16h*)&smem[W1_OFF + (n * 16 + l15) * K1P + k0 + kbB];
                    acc[n] = wmma16(a.v, b, acc[n]);
                }
            }
        }
        asm volatile("" ::: "memory");
        // h1 overlay goes into the LAST-consumed buffer (parity tc&1); the other
        // buffer is being filled by the next tile's chunk 0 right now.
        _Float16* h1w = &smem[stage_h + (tc & 1) * CHUNK_H];

        // bias + relu -> h1 (f16), layout [row][128]
        #pragma unroll
        for (int n = 0; n < 8; ++n) {
            const int col = n * 16 + l15;
            #pragma unroll
            for (int r = 0; r < 8; ++r) {
                float v = acc[n][r] + bb1[n];
                v = v > 0.0f ? v : 0.0f;
                h1w[(r + rowoff) * H1 + col] = (_Float16)v;
            }
        }

        // ---------------- layer 2: [16x128] x [128x64] ----------------
        v8f acc2[4];
        #pragma unroll
        for (int n = 0; n < 4; ++n) acc2[n] = {};
        #pragma unroll
        for (int k0 = 0; k0 < H1; k0 += 32) {
            AFrag a;
            a.h[0] = *(const v8h*)&h1w[l15 * H1 + k0 + kbB];
            a.h[1] = *(const v8h*)&h1w[l15 * H1 + k0 + kbB + 8];
            #pragma unroll
            for (int n = 0; n < 4; ++n) {
                const v16h b = *(const v16h*)&smem[W2_OFF + (n * 16 + l15) * H1 + k0 + kbB];
                acc2[n] = wmma16(a.v, b, acc2[n]);
            }
        }
        #pragma unroll
        for (int n = 0; n < 4; ++n) {
            const int col = n * 16 + l15;
            #pragma unroll
            for (int r = 0; r < 8; ++r) {
                float v = acc2[n][r] + bb2[n];
                v = v > 0.0f ? v : 0.0f;
                h2w[(r + rowoff) * H2 + col] = (_Float16)v;
            }
        }

        // ---------------- layer 3: [16x64] x [64x16(pad of 10)] ----------------
        v8f acc3 = {};
        #pragma unroll
        for (int k0 = 0; k0 < H2; k0 += 32) {
            AFrag a;
            a.h[0] = *(const v8h*)&h2w[l15 * H2 + k0 + kbB];
            a.h[1] = *(const v8h*)&h2w[l15 * H2 + k0 + kbB + 8];
            const v16h b = *(const v16h*)&smem[W3_OFF + l15 * H2 + k0 + kbB];
            acc3 = wmma16(a.v, b, acc3);
        }
        if (l15 < NOUT) {
            #pragma unroll
            for (int r = 0; r < 8; ++r) {
                out[(size_t)(m0 + r + rowoff) * NOUT + l15] = (_Float16)(acc3[r] + bb3);
            }
        }
    }
}

extern "C" void kernel_launch(void* const* d_in, const int* in_sizes, int n_in,
                              void* d_out, int out_size, void* d_ws, size_t ws_size,
                              hipStream_t stream) {
    (void)in_sizes; (void)n_in; (void)out_size; (void)d_ws; (void)ws_size;
    const _Float16* x   = (const _Float16*)d_in[0];
    const int*      w1q = (const int*)d_in[1];
    const _Float16* s1  = (const _Float16*)d_in[2];
    const _Float16* b1  = (const _Float16*)d_in[3];
    const int*      w2q = (const int*)d_in[4];
    const _Float16* s2  = (const _Float16*)d_in[5];
    const _Float16* b2  = (const _Float16*)d_in[6];
    const int*      w3q = (const int*)d_in[7];
    const _Float16* s3  = (const _Float16*)d_in[8];
    const _Float16* b3  = (const _Float16*)d_in[9];
    _Float16*       out = (_Float16*)d_out;

    const size_t shmem = (size_t)LDS_HALVES * sizeof(_Float16); // 321536 B
    qmlp_fused_w4a16<<<dim3(NWG), dim3(256), shmem, stream>>>(
        x, w1q, s1, b1, w2q, s2, b2, w3q, s3, b3, out);
}
